// LCDloss_55284819034494
// MI455X (gfx1250) — compile-verified, matrix operand order checked by reading, not run on hardware
//
#include <hip/hip_runtime.h>
#include <stdint.h>
#include <math.h>

#define NVOX (128*128*128)
#define NVOLS 4

typedef __attribute__((ext_vector_type(16))) _Float16 v16h;
typedef __attribute__((ext_vector_type(8)))  _Float16 v8h;
typedef __attribute__((ext_vector_type(8)))  float    v8f;

// workspace layout:
//   [0..8)    unsigned long long S_acc
//   [8..24)   int counts[4]
//   [128..)   int labels[4][NVOX]            (32 MiB, L2-resident)
//   then      _Float16 fg16[4][NVOX]         (16 MiB, L2-resident)
#define WS_LABELS_OFF 128
#define WS_FG_OFF     (WS_LABELS_OFF + (size_t)NVOLS*NVOX*4)

__global__ void k_zero(void* ws) {
    if (threadIdx.x == 0) *(unsigned long long*)ws = 0ull;
    int* c = (int*)((char*)ws + 8);
    if (threadIdx.x < 4) c[threadIdx.x] = 0;
}

// fused: binarize channels 1,2 of x -> fg16 (0/1 f16) and init union-find labels
__global__ void k_binarize(const float* __restrict__ x, void* ws) {
    const int groups_per_vol = NVOX / 4;
    int g = blockIdx.x * blockDim.x + threadIdx.x;
    if (g >= NVOLS * groups_per_vol) return;
    int v = g / groups_per_vol;
    int i = (g - v * groups_per_vol) * 4;
    int b = v >> 1, c = v & 1;
    const float4 xv = *(const float4*)(x + (size_t)(b * 3 + c + 1) * NVOX + i);
    int*      L  = (int*)((char*)ws + WS_LABELS_OFF) + (size_t)v * NVOX;
    _Float16* fg = (_Float16*)((char*)ws + WS_FG_OFF) + (size_t)v * NVOX;
    float f[4] = {xv.x, xv.y, xv.z, xv.w};
#pragma unroll
    for (int e = 0; e < 4; ++e) {
        bool on = f[e] > 0.0f;
        fg[i + e] = on ? (_Float16)1.0f : (_Float16)0.0f;
        L[i + e]  = on ? (i + e) : -1;
    }
}

__device__ __forceinline__ int find_root(int* L, int a) {
    int p = L[a];
    while (p != a) {
        int gp = L[p];
        if (gp != p) L[a] = gp;   // path halving (benign race)
        a = p; p = gp;
    }
    return a;
}

__device__ __forceinline__ void unite(int* L, int a, int b) {
    int ra = find_root(L, a), rb = find_root(L, b);
    while (ra != rb) {
        if (ra < rb) { int t = ra; ra = rb; rb = t; }      // hook larger onto smaller
        int old = atomicCAS(&L[ra], ra, rb);
        if (old == ra) break;
        ra = find_root(L, old);
        rb = find_root(L, rb);
    }
}

// single-pass concurrent union-find over +x/+y/+z edges (6-connectivity)
__global__ void k_merge(void* ws) {
    int g = blockIdx.x * blockDim.x + threadIdx.x;   // < 4*NVOX = 8,388,608
    int v = g >> 21;                                  // NVOX = 2^21
    int i = g & (NVOX - 1);
    int* L = (int*)((char*)ws + WS_LABELS_OFF) + (size_t)v * NVOX;
    if (L[i] == -1) return;
    int xc = i & 127, yc = (i >> 7) & 127, zc = i >> 14;
    if (xc < 127 && L[i + 1]     != -1) unite(L, i, i + 1);
    if (yc < 127 && L[i + 128]   != -1) unite(L, i, i + 128);
    if (zc < 127 && L[i + 16384] != -1) unite(L, i, i + 16384);
}

// count roots per volume (L[i]==i iff root; background is -1)
__global__ void k_count(void* ws) {
    int g = blockIdx.x * blockDim.x + threadIdx.x;
    int v = g >> 21;
    int i = g & (NVOX - 1);
    int* L = (int*)((char*)ws + WS_LABELS_OFF) + (size_t)v * NVOX;
    int flag = (L[i] == i) ? 1 : 0;
    unsigned long long m = __ballot(flag);            // wave32: low 32 bits
    int cnt = __popcll(m);
    if ((threadIdx.x & 31) == 0 && cnt)
        atomicAdd((int*)((char*)ws + 8) + v, cnt);
}

// Conv loss: 3^3 box-counts via WMMA banded GEMM (x-axis sliding sums),
// async-to-LDS plane staging, rolling 3-plane buffer, integer accumulation.
#define STG_X 144
#define STG_SLOT_BYTES (2 * 32 * STG_X * 2)   // bytes per plane slot (= 18432)
__global__ __launch_bounds__(128) void k_convloss(void* ws) {
    const int tx0  = blockIdx.x * 16;   // output-x tile base (126 outputs -> 8 tiles)
    const int ty0  = blockIdx.y * 16;
    const int b    = blockIdx.z;
    const int tid  = threadIdx.x;       // 128 threads = 4 waves (wave32)
    const int lane = tid & 31;
    const int wv   = tid >> 5;
    const int wc   = wv & 1;            // channel for this wave's WMMA
    const int wh   = wv >> 1;           // row-half (0: y rows 0..15, 1: 16..31)

    __shared__ _Float16  stage[3][2][32][STG_X];    // padded, zero-backed
    __shared__ float     sx[3][2][32][17];          // +1 pad: no bank conflicts
    __shared__ long long red[128];

    // one-time zero of stage padding (rows>=18 / x>=128 stay zero forever)
    {
        unsigned* sz = (unsigned*)&stage[0][0][0][0];
        const int nd = (int)(sizeof(stage) / 4);
#pragma unroll 4
        for (int u = tid; u < nd; u += 128) sz[u] = 0u;
    }
    __syncthreads();

    const _Float16* fg = (const _Float16*)((char*)ws + WS_FG_OFF);

    // z-invariant staging chunk descriptors (<=5 chunks of 16B per thread)
    unsigned        c_lds0[5];   // LDS offset for slot 0
    const _Float16* c_gp[5];     // global base (plane 0)
    bool            c_ok[5];
#pragma unroll
    for (int j = 0; j < 5; ++j) {
        int u  = tid + 128 * j;
        int ch = u / 288;
        int rr = (u >> 4) % 18;
        int chk = u & 15;
        int gy = ty0 + rr;
        c_ok[j]   = (u < 2 * 18 * 16) && (gy < 128);
        c_lds0[j] = (unsigned)(size_t)&stage[0][ch & 1][rr][chk * 8];
        c_gp[j]   = fg + ((size_t)(b * 2 + (ch & 1)) * NVOX +
                          (size_t)(gy & 127) * 128 + (size_t)chk * 8);
    }

    // band matrix fragment B[k][n] = 1 iff n <= k <= n+2 (ISA 16-bit B layout)
    v16h bfrag;
    {
        int n = lane & 15;
#pragma unroll
        for (int e = 0; e < 16; ++e) {
            int k = ((e < 8) ? ((lane < 16) ? 0 : 8) : ((lane < 16) ? 16 : 24)) + (e & 7);
            bfrag[e] = (_Float16)((k >= n && k <= n + 2) ? 1.0f : 0.0f);
        }
    }

    long long sacc = 0;
    // rolling slot indices kept purely in registers (no %3, no lookup table)
    int s_z = 0, s_zm1 = 2, s_zm2 = 1;

    for (int zp = 0; zp < 128; ++zp) {
        const int slot = s_z;

        // ---- async-stage plane zp (both channels, rows 0..17, x 0..127) ----
#pragma unroll
        for (int j = 0; j < 5; ++j) {
            if (c_ok[j]) {
                unsigned lds_off = c_lds0[j] + (unsigned)slot * STG_SLOT_BYTES;
                unsigned long long ga =
                    (unsigned long long)(size_t)(c_gp[j] + (size_t)zp * 16384);
                asm volatile("global_load_async_to_lds_b128 %0, %1, off"
                             :: "v"(lds_off), "v"(ga) : "memory");
            }
        }
        asm volatile("s_wait_asynccnt 0" ::: "memory");
        __syncthreads();

        // ---- one WMMA per wave: x-axis sliding sums for (channel wc, half wh) ----
        {
            const int rloc = wh * 16 + (lane & 15);           // 0..31, padded rows ok
            const int xlo  = tx0 + ((lane < 16) ? 0 : 8);     // 16B aligned
            const v8h lo = *(const v8h*)&stage[slot][wc][rloc][xlo];
            const v8h hi = *(const v8h*)&stage[slot][wc][rloc][xlo + 16];
            v16h afrag;
#pragma unroll
            for (int e = 0; e < 8; ++e) { afrag[e] = lo[e]; afrag[e + 8] = hi[e]; }
            v8f acc = {};
            acc = __builtin_amdgcn_wmma_f32_16x16x32_f16(
                false, afrag, false, bfrag, (short)0, acc, false, false);
            int n     = lane & 15;
            int mbase = wh * 16 + ((lane < 16) ? 0 : 8);
#pragma unroll
            for (int r = 0; r < 8; ++r) sx[slot][wc][mbase + r][n] = acc[r];
        }
        __syncthreads();

        // ---- accumulate loss for output plane z = zp-2 (y/z sums: plain adds) ----
        if (zp >= 2) {
            const int p0 = s_zm2, p1 = s_zm1, p2 = s_z;  // planes zp-2, zp-1, zp
            int txl = tid & 15;
            int tyb = tid >> 4;              // 0..7 ; each thread does 2 y rows
            if (tx0 + txl <= 125) {
#pragma unroll
                for (int hy = 0; hy < 2; ++hy) {
                    int tyl = tyb + hy * 8;
                    if (ty0 + tyl <= 125) {
                        float n0 = 0.0f, n1 = 0.0f;
#pragma unroll
                        for (int dy = 0; dy < 3; ++dy) {
                            n0 += sx[p0][0][tyl + dy][txl] + sx[p1][0][tyl + dy][txl]
                                + sx[p2][0][tyl + dy][txl];
                            n1 += sx[p0][1][tyl + dy][txl] + sx[p1][1][tyl + dy][txl]
                                + sx[p2][1][tyl + dy][txl];
                        }
                        int i0 = (int)(n0 + 0.5f), i1 = (int)(n1 + 0.5f);
                        sacc += (long long)((i0 > 0) ? i1 : 0)
                              + (long long)((i1 > 0) ? i0 : 0);
                    }
                }
            }
        }
        __syncthreads();
        // rotate slots: (zm2, zm1, z) <- (zm1, z, zm2)
        int t = s_zm2; s_zm2 = s_zm1; s_zm1 = s_z; s_z = t;
    }

    red[tid] = sacc;
    __syncthreads();
    for (int s = 64; s > 0; s >>= 1) {
        if (tid < s) red[tid] += red[tid + s];
        __syncthreads();
    }
    if (tid == 0)
        atomicAdd((unsigned long long*)ws, (unsigned long long)red[0]);
}

__global__ void k_finalize(const void* ws, float* out) {
    unsigned long long S = *(const unsigned long long*)ws;
    const int* counts = (const int*)((const char*)ws + 8);
    int lc = 0;
#pragma unroll
    for (int v = 0; v < 4; ++v) lc += (counts[v] > 1) ? 1 : 0;
    // lcd = lc*100 - ld/10, with ld = -S  =>  lc*100 + S/10
    float lcd = (float)lc * 100.0f + (float)S * 0.1f;
    lcd = (lcd != 0.0f) ? logf(lcd) : 0.0f;
    out[0] = lcd * 0.5f;   // / B, B == 2
}

extern "C" void kernel_launch(void* const* d_in, const int* in_sizes, int n_in,
                              void* d_out, int out_size, void* d_ws, size_t ws_size,
                              hipStream_t stream) {
    (void)in_sizes; (void)n_in; (void)out_size; (void)ws_size;
    const float* x = (const float*)d_in[0];   // d_in[1] (y) is unused by the reference
    float* out = (float*)d_out;

    hipLaunchKernelGGL(k_zero, dim3(1), dim3(64), 0, stream, d_ws);

    const int groups = NVOLS * (NVOX / 4);                 // 2,097,152
    hipLaunchKernelGGL(k_binarize, dim3(groups / 256), dim3(256), 0, stream, x, d_ws);

    const int vox_blocks = (NVOLS * NVOX) / 256;           // 32,768
    hipLaunchKernelGGL(k_merge, dim3(vox_blocks), dim3(256), 0, stream, d_ws);
    hipLaunchKernelGGL(k_count, dim3(vox_blocks), dim3(256), 0, stream, d_ws);

    hipLaunchKernelGGL(k_convloss, dim3(8, 8, 2), dim3(128), 0, stream, d_ws);
    hipLaunchKernelGGL(k_finalize, dim3(1), dim3(1), 0, stream, d_ws, out);
}
